// MultiHeadAttention_9079560863896
// MI455X (gfx1250) — compile-verified, matrix operand order checked by reading, not run on hardware
//
#include <hip/hip_runtime.h>
#include <hip/hip_bf16.h>

// Problem constants (match reference)
#define BB 8
#define LL 1024
#define DD 512
#define HH 8
#define HD 4096  // H*D

typedef __attribute__((ext_vector_type(16))) __bf16 v16bf;
typedef __attribute__((ext_vector_type(8)))  float  v8f;

__device__ __forceinline__ v8f wmma_bf16(v16bf a, v16bf b, v8f c) {
  // D = A(16x32) * B(32x16) + C(16x16), fp32 accumulate
  return __builtin_amdgcn_wmma_f32_16x16x32_bf16(
      /*neg_a=*/false, a, /*neg_b=*/false, b,
      /*c_mod=*/(short)0, c, /*reuse_a=*/false, /*reuse_b=*/false);
}

__device__ __forceinline__ __bf16 f2b(float f) { return (__bf16)f; }

// Async global->LDS copy, 16B per lane. Tracked by ASYNCcnt (in-order
// completion). LDS byte address = low 32 bits of the generic pointer
// (ISA 10.2: LDS aperture truncates addr[31:0]).
__device__ __forceinline__ void async_ld_b128(void* lds, const void* gp) {
  unsigned l = (unsigned)(size_t)lds;
  asm volatile("global_load_async_to_lds_b128 %0, %1, off"
               :: "v"(l), "v"(gp) : "memory");
}
__device__ __forceinline__ void wait_async_le4() {
  asm volatile("s_wait_asynccnt 0x4" ::: "memory");
}
__device__ __forceinline__ void wait_async_0() {
  asm volatile("s_wait_asynccnt 0x0" ::: "memory");
}

// ---------------------------------------------------------------------------
// Kernel 1: QKV projection. A(8192x512 fp32) @ W(512x4096 fp32) -> bf16
// output in per-head layout Out[((b*H+h)*L + l)*D + d], scaled by `scale`.
// Block: 256 thr, tile 128x128. grid = (N/128=32, M/128=64)
// ---------------------------------------------------------------------------
__global__ void __launch_bounds__(256)
k_proj(const float* __restrict__ A, const float* __restrict__ W,
       __bf16* __restrict__ Out, float scale) {
  __shared__ alignas(64) __bf16 sA[128 * 32];  // [m][k]
  __shared__ alignas(64) __bf16 sB[128 * 32];  // [n][k] (transposed)
  const int tid   = threadIdx.x;
  const int nBase = blockIdx.x * 128;
  const int mBase = blockIdx.y * 128;
  const int lane  = tid & 31;
  const int wave  = tid >> 5;
  const int waveM = (wave >> 1) * 32;  // 4 waves along M
  const int waveN = (wave & 1) * 64;   // 2 waves along N

  v8f acc[2][4] = {};

  for (int kc = 0; kc < DD; kc += 32) {
    // Stage A tile (128x32 fp32 -> bf16), 16 floats / thread
#pragma unroll
    for (int i = 0; i < 4; ++i) {
      int idx = tid + 256 * i;
      int row = idx >> 3, seg = idx & 7;
      const float* gp = A + (size_t)(mBase + row) * DD + kc + seg * 4;
      __builtin_prefetch(gp + 32, 0, 1);  // next K-step -> global_prefetch_b8
      float4 f = *(const float4*)gp;
      __bf16* d = sA + row * 32 + seg * 4;
      d[0] = f2b(f.x); d[1] = f2b(f.y); d[2] = f2b(f.z); d[3] = f2b(f.w);
    }
    // Stage B tile (32x128 fp32) transposed into [n][k] bf16
#pragma unroll
    for (int i = 0; i < 4; ++i) {
      int idx = tid + 256 * i;
      int k = idx >> 5, seg = idx & 31;
      float4 f = *(const float4*)(W + (size_t)(kc + k) * HD + nBase + seg * 4);
      sB[(seg * 4 + 0) * 32 + k] = f2b(f.x);
      sB[(seg * 4 + 1) * 32 + k] = f2b(f.y);
      sB[(seg * 4 + 2) * 32 + k] = f2b(f.z);
      sB[(seg * 4 + 3) * 32 + k] = f2b(f.w);
    }
    __syncthreads();

    v16bf aF[2], bF[4];
#pragma unroll
    for (int mt = 0; mt < 2; ++mt)
      aF[mt] = *(const v16bf*)(sA + (waveM + mt * 16 + (lane & 15)) * 32 + (lane >> 4) * 16);
#pragma unroll
    for (int nt = 0; nt < 4; ++nt)
      bF[nt] = *(const v16bf*)(sB + (waveN + nt * 16 + (lane & 15)) * 32 + (lane >> 4) * 16);
#pragma unroll
    for (int mt = 0; mt < 2; ++mt)
#pragma unroll
      for (int nt = 0; nt < 4; ++nt)
        acc[mt][nt] = wmma_bf16(aF[mt], bF[nt], acc[mt][nt]);
    __syncthreads();
  }

  // Epilogue: scatter into (b,h,l,d) bf16 layout
#pragma unroll
  for (int mt = 0; mt < 2; ++mt)
#pragma unroll
    for (int nt = 0; nt < 4; ++nt)
#pragma unroll
      for (int i = 0; i < 8; ++i) {
        int r = mBase + waveM + mt * 16 + (lane >> 4) * 8 + i;  // global row (b*L+l)
        int c = nBase + waveN + nt * 16 + (lane & 15);          // global col (h*D+d)
        int b = r >> 10, l = r & (LL - 1);
        int h = c >> 9,  d = c & (DD - 1);
        float v = acc[mt][nt][i] * scale;
        Out[((size_t)(b * HH + h) * LL + l) * DD + d] = f2b(v);
      }
}

// ---------------------------------------------------------------------------
// Kernel 2: scores S = qh @ kh^T per (b,h). Both operands bf16, row-major:
// B-fragment for kh^T == contiguous rows of kh. Staging is a pure bf16 copy,
// so use GLOBAL_LOAD_ASYNC_TO_LDS_B128 with double-buffered LDS and a
// one-tile software pipeline (ASYNCcnt-ordered). grid = (8, 8, 64)
// ---------------------------------------------------------------------------
__global__ void __launch_bounds__(256)
k_scores(const __bf16* __restrict__ Q, const __bf16* __restrict__ Kh,
         float* __restrict__ S) {
  __shared__ alignas(64) __bf16 sA[2][128 * 32];
  __shared__ alignas(64) __bf16 sB[2][128 * 32];
  const int tid   = threadIdx.x;
  const int bh    = blockIdx.z;
  const int nBase = blockIdx.x * 128;
  const int mBase = blockIdx.y * 128;
  const int lane  = tid & 31;
  const int wave  = tid >> 5;
  const int waveM = (wave >> 1) * 32;
  const int waveN = (wave & 1) * 64;
  const __bf16* Qb = Q  + (size_t)bh * LL * DD;
  const __bf16* Kb = Kh + (size_t)bh * LL * DD;

  v8f acc[2][4] = {};

  // 4 async instructions per thread per tile -> per-wave ASYNCcnt += 4.
  auto stage = [&](int buf, int kc) {
#pragma unroll
    for (int i = 0; i < 2; ++i) {
      int idx = tid + 256 * i;
      int row = idx >> 2, seg = idx & 3;  // 4x16B segments per 64B row
      async_ld_b128(&sA[buf][row * 32 + seg * 8],
                    Qb + (size_t)(mBase + row) * DD + kc + seg * 8);
      async_ld_b128(&sB[buf][row * 32 + seg * 8],
                    Kb + (size_t)(nBase + row) * DD + kc + seg * 8);
    }
  };

  stage(0, 0);
  for (int kc = 0; kc < DD; kc += 32) {
    const int cur = (kc >> 5) & 1;
    if (kc + 32 < DD) {
      stage(cur ^ 1, kc + 32);  // prefetch next tile, then wait for current
      wait_async_le4();         // async loads retire in order
    } else {
      wait_async_0();
    }
    __syncthreads();

    v16bf aF[2], bF[4];
#pragma unroll
    for (int mt = 0; mt < 2; ++mt)
      aF[mt] = *(const v16bf*)(&sA[cur][(waveM + mt * 16 + (lane & 15)) * 32 + (lane >> 4) * 16]);
#pragma unroll
    for (int nt = 0; nt < 4; ++nt)
      bF[nt] = *(const v16bf*)(&sB[cur][(waveN + nt * 16 + (lane & 15)) * 32 + (lane >> 4) * 16]);
#pragma unroll
    for (int mt = 0; mt < 2; ++mt)
#pragma unroll
      for (int nt = 0; nt < 4; ++nt)
        acc[mt][nt] = wmma_bf16(aF[mt], bF[nt], acc[mt][nt]);
    __syncthreads();  // protect buffer reuse two steps ahead
  }

  float* Sb = S + (size_t)bh * LL * LL;
#pragma unroll
  for (int mt = 0; mt < 2; ++mt)
#pragma unroll
    for (int nt = 0; nt < 4; ++nt)
#pragma unroll
      for (int i = 0; i < 8; ++i) {
        int r = mBase + waveM + mt * 16 + (lane >> 4) * 8 + i;
        int c = nBase + waveN + nt * 16 + (lane & 15);
        Sb[(size_t)r * LL + c] = acc[mt][nt][i];
      }
}

// ---------------------------------------------------------------------------
// Kernel 3: softmax over rows of 1024, in place on the attn output region.
// grid = 65536 rows, block = 256.
// ---------------------------------------------------------------------------
__global__ void __launch_bounds__(256)
k_softmax(float* __restrict__ S) {
  __shared__ float red[256];
  const int tid = threadIdx.x;
  float* row = S + (size_t)blockIdx.x * LL;

  float x[4];
  float m = -3.4e38f;
#pragma unroll
  for (int j = 0; j < 4; ++j) { x[j] = row[tid + 256 * j]; m = fmaxf(m, x[j]); }
  red[tid] = m;
  __syncthreads();
  for (int s = 128; s > 0; s >>= 1) {
    if (tid < s) red[tid] = fmaxf(red[tid], red[tid + s]);
    __syncthreads();
  }
  m = red[0];
  __syncthreads();

  float sum = 0.f;
#pragma unroll
  for (int j = 0; j < 4; ++j) { x[j] = __expf(x[j] - m); sum += x[j]; }
  red[tid] = sum;
  __syncthreads();
  for (int s = 128; s > 0; s >>= 1) {
    if (tid < s) red[tid] += red[tid + s];
    __syncthreads();
  }
  float inv = 1.0f / red[0];
#pragma unroll
  for (int j = 0; j < 4; ++j) row[tid + 256 * j] = x[j] * inv;
}

// ---------------------------------------------------------------------------
// Kernel 4: ctx = attn(fp32) @ vh(bf16) per (b,h); output bf16 in (B,L,H*D).
// grid = (4, 8, 64)
// ---------------------------------------------------------------------------
__global__ void __launch_bounds__(256)
k_ctx(const float* __restrict__ Attn, const __bf16* __restrict__ V,
      __bf16* __restrict__ Ctx) {
  __shared__ alignas(64) __bf16 sA[128 * 32];
  __shared__ alignas(64) __bf16 sB[128 * 32];  // [n][k]
  const int tid   = threadIdx.x;
  const int bh    = blockIdx.z;
  const int nBase = blockIdx.x * 128;
  const int mBase = blockIdx.y * 128;
  const int lane  = tid & 31;
  const int wave  = tid >> 5;
  const int waveM = (wave >> 1) * 32;
  const int waveN = (wave & 1) * 64;
  const float*  Ab = Attn + (size_t)bh * LL * LL;
  const __bf16* Vb = V    + (size_t)bh * LL * DD;

  v8f acc[2][4] = {};

  for (int kc = 0; kc < LL; kc += 32) {
    // A: 128x32 fp32 attn -> bf16
#pragma unroll
    for (int i = 0; i < 4; ++i) {
      int idx = tid + 256 * i;
      int row = idx >> 3, seg = idx & 7;
      const float* gp = Ab + (size_t)(mBase + row) * LL + kc + seg * 4;
      __builtin_prefetch(gp + 32, 0, 1);
      float4 f = *(const float4*)gp;
      __bf16* d = sA + row * 32 + seg * 4;
      d[0] = f2b(f.x); d[1] = f2b(f.y); d[2] = f2b(f.z); d[3] = f2b(f.w);
    }
    // B: vh 32(k) x 128(n) bf16, transpose into [n][k]
#pragma unroll
    for (int i = 0; i < 2; ++i) {
      int idx = tid + 256 * i;
      int k = idx >> 4, seg = idx & 15;
      uint4 u = *(const uint4*)(Vb + (size_t)(kc + k) * DD + nBase + seg * 8);
      alignas(16) __bf16 t[8];
      *(uint4*)t = u;
#pragma unroll
      for (int j = 0; j < 8; ++j) sB[(seg * 8 + j) * 32 + k] = t[j];
    }
    __syncthreads();

    v16bf aF[2], bF[4];
#pragma unroll
    for (int mt = 0; mt < 2; ++mt)
      aF[mt] = *(const v16bf*)(sA + (waveM + mt * 16 + (lane & 15)) * 32 + (lane >> 4) * 16);
#pragma unroll
    for (int nt = 0; nt < 4; ++nt)
      bF[nt] = *(const v16bf*)(sB + (waveN + nt * 16 + (lane & 15)) * 32 + (lane >> 4) * 16);
#pragma unroll
    for (int mt = 0; mt < 2; ++mt)
#pragma unroll
      for (int nt = 0; nt < 4; ++nt)
        acc[mt][nt] = wmma_bf16(aF[mt], bF[nt], acc[mt][nt]);
    __syncthreads();
  }

  const int b = bh >> 3, h = bh & 7;
#pragma unroll
  for (int mt = 0; mt < 2; ++mt)
#pragma unroll
    for (int nt = 0; nt < 4; ++nt)
#pragma unroll
      for (int i = 0; i < 8; ++i) {
        int l = mBase + waveM + mt * 16 + (lane >> 4) * 8 + i;
        int d = nBase + waveN + nt * 16 + (lane & 15);
        Ctx[((size_t)(b * LL + l) * HD) + h * DD + d] = f2b(acc[mt][nt][i]);
      }
}

// ---------------------------------------------------------------------------
// Kernel 5: out_pre = ctx(8192x4096 bf16) @ w_fc(4096x512 fp32) + residual.
// A tile is a pure bf16 copy -> async global->LDS, overlapped with the
// fp32->bf16 VALU conversion of the weight tile. grid = (4, 64)
// ---------------------------------------------------------------------------
__global__ void __launch_bounds__(256)
k_fc(const __bf16* __restrict__ Ctx, const float* __restrict__ Wfc,
     const float* __restrict__ Res, float* __restrict__ OutP) {
  __shared__ alignas(64) __bf16 sA[128 * 32];
  __shared__ alignas(64) __bf16 sB[128 * 32];
  const int tid   = threadIdx.x;
  const int nBase = blockIdx.x * 128;
  const int mBase = blockIdx.y * 128;
  const int lane  = tid & 31;
  const int wave  = tid >> 5;
  const int waveM = (wave >> 1) * 32;
  const int waveN = (wave & 1) * 64;

  v8f acc[2][4] = {};

  for (int kc = 0; kc < HD; kc += 32) {
    // A tile via async copies (in flight while we convert B below)
#pragma unroll
    for (int i = 0; i < 2; ++i) {
      int idx = tid + 256 * i;
      int row = idx >> 2, seg = idx & 3;
      async_ld_b128(sA + row * 32 + seg * 8,
                    Ctx + (size_t)(mBase + row) * HD + kc + seg * 8);
    }
    // B tile: w_fc 32(k) x 128(n) fp32 -> bf16, transposed [n][k]
#pragma unroll
    for (int i = 0; i < 4; ++i) {
      int idx = tid + 256 * i;
      int k = idx >> 5, seg = idx & 31;
      float4 f = *(const float4*)(Wfc + (size_t)(kc + k) * DD + nBase + seg * 4);
      sB[(seg * 4 + 0) * 32 + k] = f2b(f.x);
      sB[(seg * 4 + 1) * 32 + k] = f2b(f.y);
      sB[(seg * 4 + 2) * 32 + k] = f2b(f.z);
      sB[(seg * 4 + 3) * 32 + k] = f2b(f.w);
    }
    wait_async_0();
    __syncthreads();

    v16bf aF[2], bF[4];
#pragma unroll
    for (int mt = 0; mt < 2; ++mt)
      aF[mt] = *(const v16bf*)(sA + (waveM + mt * 16 + (lane & 15)) * 32 + (lane >> 4) * 16);
#pragma unroll
    for (int nt = 0; nt < 4; ++nt)
      bF[nt] = *(const v16bf*)(sB + (waveN + nt * 16 + (lane & 15)) * 32 + (lane >> 4) * 16);
#pragma unroll
    for (int mt = 0; mt < 2; ++mt)
#pragma unroll
      for (int nt = 0; nt < 4; ++nt)
        acc[mt][nt] = wmma_bf16(aF[mt], bF[nt], acc[mt][nt]);
    __syncthreads();
  }

#pragma unroll
  for (int mt = 0; mt < 2; ++mt)
#pragma unroll
    for (int nt = 0; nt < 4; ++nt)
#pragma unroll
      for (int i = 0; i < 8; ++i) {
        int r = mBase + waveM + mt * 16 + (lane >> 4) * 8 + i;
        int c = nBase + waveN + nt * 16 + (lane & 15);
        OutP[(size_t)r * DD + c] = acc[mt][nt][i] + Res[(size_t)r * DD + c];
      }
}

// ---------------------------------------------------------------------------
// Kernel 6: LayerNorm over rows of 512. grid = 8192, block = 256.
// ---------------------------------------------------------------------------
__global__ void __launch_bounds__(256)
k_ln(const float* __restrict__ X, const float* __restrict__ g,
     const float* __restrict__ be, float* __restrict__ Out) {
  __shared__ float r1[256];
  __shared__ float r2[256];
  const int tid = threadIdx.x;
  const size_t base = (size_t)blockIdx.x * DD;
  float x0 = X[base + tid];
  float x1 = X[base + tid + 256];
  r1[tid] = x0 + x1;
  r2[tid] = x0 * x0 + x1 * x1;
  __syncthreads();
  for (int s = 128; s > 0; s >>= 1) {
    if (tid < s) { r1[tid] += r1[tid + s]; r2[tid] += r2[tid + s]; }
    __syncthreads();
  }
  float mean = r1[0] * (1.0f / DD);
  float var  = r2[0] * (1.0f / DD) - mean * mean;
  float inv  = rsqrtf(var + 1e-6f);
  Out[base + tid]       = (x0 - mean) * inv * g[tid]       + be[tid];
  Out[base + tid + 256] = (x1 - mean) * inv * g[tid + 256] + be[tid + 256];
}

// ---------------------------------------------------------------------------
extern "C" void kernel_launch(void* const* d_in, const int* in_sizes, int n_in,
                              void* d_out, int out_size, void* d_ws, size_t ws_size,
                              hipStream_t stream) {
  const float* q     = (const float*)d_in[0];
  const float* k     = (const float*)d_in[1];
  const float* v     = (const float*)d_in[2];
  const float* wq    = (const float*)d_in[3];
  const float* wk    = (const float*)d_in[4];
  const float* wv    = (const float*)d_in[5];
  const float* wfc   = (const float*)d_in[6];
  const float* gamma = (const float*)d_in[7];
  const float* beta  = (const float*)d_in[8];

  char* ws = (char*)d_ws;
  __bf16* qh   = (__bf16*)(ws);                         //  64 MB
  __bf16* kh   = (__bf16*)(ws + (size_t)67108864);      //  64 MB
  __bf16* vh   = (__bf16*)(ws + (size_t)134217728);     //  64 MB
  __bf16* ctxp = (__bf16*)(ws + (size_t)201326592);     //  64 MB
  float*  outp = (float* )(ws + (size_t)268435456);     //  16 MB

  float* out  = (float*)d_out;                          // (B,L,D) fp32
  float* attn = out + (size_t)BB * LL * DD;             // (B,H,L,L) fp32

  const dim3 blk(256);
  const float qscale = 0.04419417382415922f;            // 1/sqrt(512)

  k_proj<<<dim3(32, 64), blk, 0, stream>>>(q, wq, qh, qscale);
  k_proj<<<dim3(32, 64), blk, 0, stream>>>(k, wk, kh, 1.0f);
  k_proj<<<dim3(32, 64), blk, 0, stream>>>(v, wv, vh, 1.0f);
  k_scores<<<dim3(8, 8, 64), blk, 0, stream>>>(qh, kh, attn);
  k_softmax<<<dim3(64 * 1024), blk, 0, stream>>>(attn);
  k_ctx<<<dim3(4, 8, 64), blk, 0, stream>>>(attn, vh, ctxp);
  k_fc<<<dim3(4, 64), blk, 0, stream>>>(ctxp, wfc, q, outp);
  k_ln<<<dim3(8192), blk, 0, stream>>>(outp, gamma, beta, out);
}
